// CrossAttentionModule_69681549410847
// MI455X (gfx1250) — matrix operand order repaired and checked by reading, not compiled, not run on hardware
//
#include <hip/hip_runtime.h>

typedef _Float16 v16h __attribute__((ext_vector_type(16)));
typedef _Float16 v8h  __attribute__((ext_vector_type(8)));
typedef float    v8f  __attribute__((ext_vector_type(8)));
typedef float    v4f  __attribute__((ext_vector_type(4)));

constexpr int BATCH = 8;
constexpr int SEQ   = 2048;
constexpr int DIM   = 768;
constexpr float SM_SCALE = 0.03608439182435161f; // 1/sqrt(768)

// Staged-GEMM geometry (scores & pv kernels)
constexpr int MBLK = 128;               // A rows per block
constexpr int NBLK = 64;                // B rows per block
constexpr int KC   = 64;                // contraction chunk (f16 elems)
constexpr int LDS_STRIDE = KC * 2 + 16; // 144 B row stride (36 dw -> conflict-light)
constexpr int A_BYTES = MBLK * LDS_STRIDE;  // 18432
constexpr int B_BYTES = NBLK * LDS_STRIDE;  // 9216
constexpr int BUF_BYTES = A_BYTES + B_BYTES; // 27648

// ---------------------------------------------------------------------------
// WMMA helper: D = A(16x32 f16) * B(32x16 f16) + C(16x16 f32)
// ---------------------------------------------------------------------------
__device__ __forceinline__ v8f wmma_f16(v16h a, v16h b, v8f c) {
    return __builtin_amdgcn_wmma_f32_16x16x32_f16(
        false, a, false, b, (short)0, c, false, false);
}

// A operand (16x32 f16): lane row = lane%16; half h = lane>>4 holds K chunks
// [h*8 .. h*8+7] and [16+h*8 .. 16+h*8+7].
__device__ __forceinline__ v16h load_a_f16(const _Float16* row, int kb, int h) {
    v8h a0 = *(const v8h*)(row + kb + h * 8);
    v8h a1 = *(const v8h*)(row + kb + 16 + h * 8);
    return __builtin_shufflevector(a0, a1, 0,1,2,3,4,5,6,7,8,9,10,11,12,13,14,15);
}
// B operand (32x16 f16): lane col = lane%16; half h holds K chunks
// [h*8 .. h*8+7] and [16+h*8 .. 16+h*8+7].
__device__ __forceinline__ v16h load_b_f16(const _Float16* row, int kb, int h) {
    v8h b0 = *(const v8h*)(row + kb + h * 8);
    v8h b1 = *(const v8h*)(row + kb + 16 + h * 8);
    return __builtin_shufflevector(b0, b1, 0,1,2,3,4,5,6,7,8,9,10,11,12,13,14,15);
}

// f32 -> f16 converting loaders (projection kernel)
__device__ __forceinline__ v16h load_a_f32(const float* row, int kb, int h) {
    v4f a0 = *(const v4f*)(row + kb + h * 8);
    v4f a1 = *(const v4f*)(row + kb + h * 8 + 4);
    v4f a2 = *(const v4f*)(row + kb + 16 + h * 8);
    v4f a3 = *(const v4f*)(row + kb + 16 + h * 8 + 4);
    v16h a;
    #pragma unroll
    for (int i = 0; i < 4; ++i) {
        a[i] = (_Float16)a0[i];      a[4 + i]  = (_Float16)a1[i];
        a[8 + i] = (_Float16)a2[i];  a[12 + i] = (_Float16)a3[i];
    }
    return a;
}
__device__ __forceinline__ v16h load_b_f32(const float* row, int kb, int h) {
    v4f w0 = *(const v4f*)(row + kb + h * 8);
    v4f w1 = *(const v4f*)(row + kb + h * 8 + 4);
    v4f w2 = *(const v4f*)(row + kb + 16 + h * 8);
    v4f w3 = *(const v4f*)(row + kb + 16 + h * 8 + 4);
    v16h b;
    #pragma unroll
    for (int i = 0; i < 4; ++i) {
        b[i] = (_Float16)w0[i];      b[4 + i]  = (_Float16)w1[i];
        b[8 + i] = (_Float16)w2[i];  b[12 + i] = (_Float16)w3[i];
    }
    return b;
}

// ---------------------------------------------------------------------------
// gfx1250 async global->LDS copy (16B per lane), tracked by ASYNCcnt.
// ---------------------------------------------------------------------------
__device__ __forceinline__ void async_cp16(void* lds, const void* gaddr) {
    unsigned lo = (unsigned)(uintptr_t)lds;     // low 32 bits = LDS byte offset
    asm volatile("global_load_async_to_lds_b128 %0, %1, off"
                 :: "v"(lo), "v"(gaddr) : "memory");
}
__device__ __forceinline__ void async_wait0() {
    asm volatile("s_wait_asynccnt 0x0" ::: "memory");
}

// Stage one KC-deep chunk of A (MBLK rows) and B (NBLK rows) into LDS.
// 256 threads, 6 async b128 ops per thread.
__device__ __forceinline__ void stage_chunk(
    const _Float16* __restrict__ A_rows, const _Float16* __restrict__ B_rows,
    int kd, int kb, char* bufA, char* bufB, int t)
{
    {   // A: 128 rows x 128B; 2 threads/row, 64B (4 x b128) each
        const int row = t & 127, half = t >> 7;
        const _Float16* g = A_rows + (size_t)row * kd + kb + half * 32;
        char* l = bufA + row * LDS_STRIDE + half * 64;
        #pragma unroll
        for (int i = 0; i < 4; ++i) async_cp16(l + i * 16, g + i * 8);
    }
    {   // B: 64 rows x 128B; 4 threads/row, 32B (2 x b128) each
        const int row = t & 63, q4 = t >> 6;
        const _Float16* g = B_rows + (size_t)row * kd + kb + q4 * 16;
        char* l = bufB + row * LDS_STRIDE + q4 * 32;
        #pragma unroll
        for (int i = 0; i < 2; ++i) async_cp16(l + i * 16, g + i * 8);
    }
}

// Wave computes its 32x32 tile from the staged chunk: 2 k-steps x 4 WMMAs.
__device__ __forceinline__ void compute_chunk(
    const char* bufA, const char* bufB, int wm, int wn, int h, int q, v8f c[4])
{
    const _Float16* ar0 = (const _Float16*)(bufA + (wm * 32 + q) * LDS_STRIDE);
    const _Float16* ar1 = (const _Float16*)(bufA + (wm * 32 + 16 + q) * LDS_STRIDE);
    const _Float16* br0 = (const _Float16*)(bufB + (wn * 32 + q) * LDS_STRIDE);
    const _Float16* br1 = (const _Float16*)(bufB + (wn * 32 + 16 + q) * LDS_STRIDE);
    #pragma unroll
    for (int ks = 0; ks < KC / 32; ++ks) {
        const int kb = ks * 32;
        v16h a0 = load_a_f16(ar0, kb, h);
        v16h a1 = load_a_f16(ar1, kb, h);
        v16h b0 = load_b_f16(br0, kb, h);
        v16h b1 = load_b_f16(br1, kb, h);
        c[0] = wmma_f16(a0, b0, c[0]);
        c[1] = wmma_f16(a0, b1, c[1]);
        c[2] = wmma_f16(a1, b0, c[2]);
        c[3] = wmma_f16(a1, b1, c[3]);
    }
}

// Double-buffered staged main loop shared by scores/pv.
__device__ __forceinline__ void gemm_mainloop(
    const _Float16* A_rows, const _Float16* B_rows, int kd, int nchunks,
    char* smem, int t, int wm, int wn, int h, int q, v8f c[4])
{
    stage_chunk(A_rows, B_rows, kd, 0, smem, smem + A_BYTES, t);
    for (int ch = 0; ch < nchunks; ++ch) {
        async_wait0();
        __syncthreads();
        if (ch + 1 < nchunks) {
            char* nb = smem + ((ch + 1) & 1) * BUF_BYTES;
            stage_chunk(A_rows, B_rows, kd, (ch + 1) * KC, nb, nb + A_BYTES, t);
        }
        const char* cb = smem + (ch & 1) * BUF_BYTES;
        compute_chunk(cb, cb + A_BYTES, wm, wn, h, q, c);
    }
}

// ---------------------------------------------------------------------------
// Kernel 1: fused QKV projection, 2x4 blocked (32 rows x 64 cols per wave).
//   Q = video @ Wq^T + bq   (f16 row-major [B*L][D])
//   K = audio @ Wk^T + bk   (f16 row-major [B*L][D])
//   V = audio @ Wv^T + bv   stored TRANSPOSED: Vt[b][d][key] (f16)
// ---------------------------------------------------------------------------
__global__ __launch_bounds__(256) void proj_kernel(
    const float* __restrict__ video, const float* __restrict__ audio,
    const float* __restrict__ Wq, const float* __restrict__ bq,
    const float* __restrict__ Wk, const float* __restrict__ bk,
    const float* __restrict__ Wv, const float* __restrict__ bv,
    _Float16* __restrict__ Qh, _Float16* __restrict__ Kh,
    _Float16* __restrict__ Vt)
{
    const int lane = threadIdx.x & 31;
    const int wave = threadIdx.x >> 5;
    const int gid  = blockIdx.x * 8 + wave;

    const int N4 = DIM / 64;             // 12
    const int M2 = (BATCH * SEQ) / 32;   // 512
    const int which = gid / (M2 * N4);   // 0=Q 1=K 2=V
    const int rem   = gid % (M2 * N4);
    const int m0 = (rem / N4) * 32;
    const int n0 = (rem % N4) * 64;

    const int h = lane >> 4;
    const int q = lane & 15;

    const float* in   = (which == 0) ? video : audio;
    const float* W    = (which == 0) ? Wq : (which == 1) ? Wk : Wv;
    const float* bias = (which == 0) ? bq : (which == 1) ? bk : bv;

    const float* arow0 = in + (size_t)(m0 + q) * DIM;
    const float* arow1 = in + (size_t)(m0 + 16 + q) * DIM;
    const float* brow[4];
    #pragma unroll
    for (int jb = 0; jb < 4; ++jb)
        brow[jb] = W + (size_t)(n0 + jb * 16 + q) * DIM;

    v8f c0[4] = {}, c1[4] = {};
    for (int kt = 0; kt < DIM / 32; ++kt) {
        const int kb = kt * 32;
        __builtin_prefetch(arow0 + kb + 128, 0, 0);
        v16h a0 = load_a_f32(arow0, kb, h);
        v16h a1 = load_a_f32(arow1, kb, h);
        #pragma unroll
        for (int jb = 0; jb < 4; ++jb) {
            v16h b = load_b_f32(brow[jb], kb, h);
            c0[jb] = wmma_f16(a0, b, c0[jb]);
            c1[jb] = wmma_f16(a1, b, c1[jb]);
        }
    }

    #pragma unroll
    for (int jb = 0; jb < 4; ++jb) {
        const float bv_ = bias[n0 + jb * 16 + q];
        #pragma unroll
        for (int j = 0; j < 8; ++j) { c0[jb][j] += bv_; c1[jb][j] += bv_; }
    }

    if (which < 2) {
        _Float16* out = (which == 0) ? Qh : Kh;
        #pragma unroll
        for (int jb = 0; jb < 4; ++jb)
            #pragma unroll
            for (int j = 0; j < 8; ++j) {
                out[(size_t)(m0 + j + 8 * h) * DIM + n0 + jb * 16 + q]      = (_Float16)c0[jb][j];
                out[(size_t)(m0 + 16 + j + 8 * h) * DIM + n0 + jb * 16 + q] = (_Float16)c1[jb][j];
            }
    } else {
        const int b_idx = m0 / SEQ;
        _Float16* vb = Vt + (size_t)b_idx * DIM * SEQ;
        const int key0 = (m0 % SEQ) + 8 * h;
        #pragma unroll
        for (int jb = 0; jb < 4; ++jb) {
            v8h p0, p1;
            #pragma unroll
            for (int j = 0; j < 8; ++j) { p0[j] = (_Float16)c0[jb][j]; p1[j] = (_Float16)c1[jb][j]; }
            *(v8h*)(vb + (size_t)(n0 + jb * 16 + q) * SEQ + key0)      = p0;
            *(v8h*)(vb + (size_t)(n0 + jb * 16 + q) * SEQ + key0 + 16) = p1;
        }
    }
}

// ---------------------------------------------------------------------------
// Kernel 2: S = (Q . K^T) * 1/sqrt(D). Block computes 128(q) x 64(k) via the
// async-LDS staged mainloop; waves in 4x2 grid of 32x32 tiles.
// ---------------------------------------------------------------------------
__global__ __launch_bounds__(256) void scores_kernel(
    const _Float16* __restrict__ Qh, const _Float16* __restrict__ Kh,
    float* __restrict__ S)
{
    __shared__ __align__(16) char smem[2 * BUF_BYTES];
    const int t    = threadIdx.x;
    const int lane = t & 31;
    const int wave = t >> 5;
    const int wm = wave >> 1, wn = wave & 1;
    const int h = lane >> 4, q = lane & 15;

    const int QB = SEQ / MBLK;  // 16
    const int KB = SEQ / NBLK;  // 32
    const int b    = blockIdx.x / (QB * KB);
    const int rem  = blockIdx.x % (QB * KB);
    const int qbase = (rem / KB) * MBLK;
    const int kbase = (rem % KB) * NBLK;

    const _Float16* A_rows = Qh + ((size_t)b * SEQ + qbase) * DIM;
    const _Float16* B_rows = Kh + ((size_t)b * SEQ + kbase) * DIM;

    v8f c[4] = {};
    gemm_mainloop(A_rows, B_rows, DIM, DIM / KC, smem, t, wm, wn, h, q, c);

    float* srow = S + (size_t)b * SEQ * SEQ;
    const int q0 = qbase + wm * 32;
    const int k0 = kbase + wn * 32;
    #pragma unroll
    for (int j = 0; j < 8; ++j) {
        srow[(size_t)(q0 + j + 8 * h) * SEQ + k0 + q]           = c[0][j] * SM_SCALE;
        srow[(size_t)(q0 + j + 8 * h) * SEQ + k0 + 16 + q]      = c[1][j] * SM_SCALE;
        srow[(size_t)(q0 + 16 + j + 8 * h) * SEQ + k0 + q]      = c[2][j] * SM_SCALE;
        srow[(size_t)(q0 + 16 + j + 8 * h) * SEQ + k0 + 16 + q] = c[3][j] * SM_SCALE;
    }
}

// ---------------------------------------------------------------------------
// Kernel 3: row softmax over 2048 keys. One wave per row, registers + shfl.
// ---------------------------------------------------------------------------
__global__ __launch_bounds__(256) void softmax_kernel(
    const float* __restrict__ S, _Float16* __restrict__ P)
{
    const int lane = threadIdx.x & 31;
    const int wave = threadIdx.x >> 5;
    const int row  = blockIdx.x * 8 + wave;

    const float* s = S + (size_t)row * SEQ;
    float x[64];
    float m = -3.4e38f;
    #pragma unroll
    for (int i = 0; i < 64; ++i) { x[i] = s[lane + i * 32]; m = fmaxf(m, x[i]); }
    #pragma unroll
    for (int off = 16; off >= 1; off >>= 1) m = fmaxf(m, __shfl_xor(m, off, 32));

    float sum = 0.f;
    #pragma unroll
    for (int i = 0; i < 64; ++i) { x[i] = __expf(x[i] - m); sum += x[i]; }
    #pragma unroll
    for (int off = 16; off >= 1; off >>= 1) sum += __shfl_xor(sum, off, 32);
    const float inv = 1.0f / sum;

    _Float16* p = P + (size_t)row * SEQ;
    #pragma unroll
    for (int i = 0; i < 64; ++i) p[lane + i * 32] = (_Float16)(x[i] * inv);
}

// ---------------------------------------------------------------------------
// Kernel 4: O = P . V. Same staged block GEMM; contraction = 2048 keys.
// A = P rows (f16 contiguous in k); B rows = Vt[d][key] (contiguous in key).
// ---------------------------------------------------------------------------
__global__ __launch_bounds__(256) void pv_kernel(
    const _Float16* __restrict__ P, const _Float16* __restrict__ Vt,
    float* __restrict__ out)
{
    __shared__ __align__(16) char smem[2 * BUF_BYTES];
    const int t    = threadIdx.x;
    const int lane = t & 31;
    const int wave = t >> 5;
    const int wm = wave >> 1, wn = wave & 1;
    const int h = lane >> 4, q = lane & 15;

    const int QB = SEQ / MBLK;  // 16
    const int NB = DIM / NBLK;  // 12
    const int b    = blockIdx.x / (QB * NB);
    const int rem  = blockIdx.x % (QB * NB);
    const int qbase = (rem / NB) * MBLK;
    const int nbase = (rem % NB) * NBLK;

    const _Float16* A_rows = P  + ((size_t)b * SEQ + qbase) * SEQ;
    const _Float16* B_rows = Vt + ((size_t)b * DIM + nbase) * SEQ;

    v8f c[4] = {};
    gemm_mainloop(A_rows, B_rows, SEQ, SEQ / KC, smem, t, wm, wn, h, q, c);

    float* ob = out + (size_t)b * SEQ * DIM;
    const int q0 = qbase + wm * 32;
    const int n0 = nbase + wn * 32;
    #pragma unroll
    for (int j = 0; j < 8; ++j) {
        ob[(size_t)(q0 + j + 8 * h) * DIM + n0 + q]           = c[0][j];
        ob[(size_t)(q0 + j + 8 * h) * DIM + n0 + 16 + q]      = c[1][j];
        ob[(size_t)(q0 + 16 + j + 8 * h) * DIM + n0 + q]      = c[2][j];
        ob[(size_t)(q0 + 16 + j + 8 * h) * DIM + n0 + 16 + q] = c[3][j];
    }
}

// ---------------------------------------------------------------------------
extern "C" void kernel_launch(void* const* d_in, const int* in_sizes, int n_in,
                              void* d_out, int out_size, void* d_ws, size_t ws_size,
                              hipStream_t stream) {
    (void)in_sizes; (void)n_in; (void)out_size; (void)ws_size;

    const float* video = (const float*)d_in[0];
    const float* audio = (const float*)d_in[1];
    const float* Wq = (const float*)d_in[2];
    const float* bq = (const float*)d_in[3];
    const float* Wk = (const float*)d_in[4];
    const float* bk = (const float*)d_in[5];
    const float* Wv = (const float*)d_in[6];
    const float* bv = (const float*)d_in[7];
    float* out = (float*)d_out;

    char* ws = (char*)d_ws;
    const size_t nAct = (size_t)BATCH * SEQ * DIM;
    _Float16* Qh = (_Float16*)ws; ws += nAct * sizeof(_Float16);
    _Float16* Kh = (_Float16*)ws; ws += nAct * sizeof(_Float16);
    _Float16* Vt = (_Float16*)ws; ws += nAct * sizeof(_Float16);
    float*    S  = (float*)ws;    ws += (size_t)BATCH * SEQ * SEQ * sizeof(float);
    _Float16* P  = (_Float16*)ws;

    // proj: 3 * 512 * 12 = 18432 wave-tiles, 8 waves/block
    proj_kernel<<<2304, 256, 0, stream>>>(video, audio, Wq, bq, Wk, bk, Wv, bv,
                                          Qh, Kh, Vt);
    // scores: 8 * 16 * 32 = 4096 blocks (128x64 tile per block)
    scores_kernel<<<4096, 256, 0, stream>>>(Qh, Kh, S);
    // softmax: B*SEQ = 16384 rows, 1 wave per row
    softmax_kernel<<<2048, 256, 0, stream>>>(S, P);
    // pv: 8 * 16 * 12 = 1536 blocks
    pv_kernel<<<1536, 256, 0, stream>>>(P, Vt, out);
}
